// QuantizedLinear_64330020159885
// MI455X (gfx1250) — matrix operand order repaired and checked by reading, not compiled, not run on hardware
//
#include <hip/hip_runtime.h>

typedef __attribute__((ext_vector_type(16))) __bf16 v16bf;
typedef __attribute__((ext_vector_type(8)))  __bf16 v8bf;
typedef __attribute__((ext_vector_type(8)))  float  v8f;

#define GROUPSZ 64
#define TM 128      // block tile rows (M)
#define TO 128      // block tile cols (K-out)
#define TK 64       // reduction step per stage (2 wmma k-slices)
#define LDSS 72     // LDS row stride in bf16 (144B: 16B-aligned, bank-conflict-free)

// ---------------- Kernel 1: dequantize W_q -> bf16 ----------------
__global__ __launch_bounds__(256) void dequant_w_bf16(
    const int* __restrict__ Wq, const float* __restrict__ scales,
    const float* __restrict__ zeros, const int* __restrict__ mask,
    const float* __restrict__ scale2, __bf16* __restrict__ out,
    int N, int G, long long total8) {
  long long idx = (long long)blockIdx.x * blockDim.x + threadIdx.x;
  if (idx >= total8) return;
  int n8 = N >> 3;
  int o  = (int)(idx / n8);
  int n0 = ((int)(idx % n8)) << 3;          // 8 elems never cross a group of 64
  int g  = n0 >> 6;
  float s = scales[(size_t)o * G + g] * scale2[o];
  float z = zeros [(size_t)o * G + g];
  size_t base = (size_t)o * N + n0;
  int4 q0 = *(const int4*)(Wq + base);
  int4 q1 = *(const int4*)(Wq + base + 4);
  int4 m0 = *(const int4*)(mask + base);
  int4 m1 = *(const int4*)(mask + base + 4);
  v8bf r;
  r[0] = (__bf16)(((float)q0.x - z) * s * (float)m0.x);
  r[1] = (__bf16)(((float)q0.y - z) * s * (float)m0.y);
  r[2] = (__bf16)(((float)q0.z - z) * s * (float)m0.z);
  r[3] = (__bf16)(((float)q0.w - z) * s * (float)m0.w);
  r[4] = (__bf16)(((float)q1.x - z) * s * (float)m1.x);
  r[5] = (__bf16)(((float)q1.y - z) * s * (float)m1.y);
  r[6] = (__bf16)(((float)q1.z - z) * s * (float)m1.z);
  r[7] = (__bf16)(((float)q1.w - z) * s * (float)m1.w);
  *(v8bf*)(out + base) = r;
}

// ---------------- Kernel 2: x f32 -> bf16 ----------------
__global__ __launch_bounds__(256) void cvt_x_bf16(
    const float* __restrict__ x, __bf16* __restrict__ out, long long total8) {
  long long idx = (long long)blockIdx.x * blockDim.x + threadIdx.x;
  if (idx >= total8) return;
  size_t base = (size_t)idx << 3;
  float4 a = *(const float4*)(x + base);
  float4 b = *(const float4*)(x + base + 4);
  v8bf r;
  r[0] = (__bf16)a.x; r[1] = (__bf16)a.y; r[2] = (__bf16)a.z; r[3] = (__bf16)a.w;
  r[4] = (__bf16)b.x; r[5] = (__bf16)b.y; r[6] = (__bf16)b.z; r[7] = (__bf16)b.w;
  *(v8bf*)(out + base) = r;
}

// Issue 4 consecutive 16B async global->LDS copies (64B half-row per lane).
// INST_OFFSET is added to BOTH the LDS and global addresses, so one LDS addr
// VGPR + one 32-bit global offset VGPR + a uniform SGPR base cover all four.
__device__ __forceinline__ void async_half_row(unsigned lds, unsigned goff,
                                               const void* sbase) {
  asm volatile(
      "global_load_async_to_lds_b128 %0, %1, %2 offset:0\n\t"
      "global_load_async_to_lds_b128 %0, %1, %2 offset:16\n\t"
      "global_load_async_to_lds_b128 %0, %1, %2 offset:32\n\t"
      "global_load_async_to_lds_b128 %0, %1, %2 offset:48"
      :: "v"(lds), "v"(goff), "s"(sbase) : "memory");
}

__device__ __forceinline__ void wait_async_all() {
  asm volatile("s_wait_asynccnt 0x0" ::: "memory");
}

// ---------------- Kernel 3: GEMM out = x_bf16 @ W_bf16^T + bias ----------------
// 256 threads = 8 wave32, wave grid 4(M) x 2(O), per-wave 32x64 output via
// 2x4 v8f accumulators. Double-buffered LDS tiles filled by the async
// global->LDS path (ASYNCcnt) while WMMAs consume the other buffer.
__global__ __launch_bounds__(256) void gemm_bf16_wmma(
    const __bf16* __restrict__ xb, const __bf16* __restrict__ wb,
    const float* __restrict__ bias, float* __restrict__ out,
    int N, int Kout) {
  __shared__ __bf16 sA[2][TM * LDSS];
  __shared__ __bf16 sB[2][TO * LDSS];

  const int tid  = threadIdx.x;
  const int lane = tid & 31;
  const int wid  = tid >> 5;
  const int wm   = wid & 3;       // wave row-tile (32 rows each)
  const int wo   = wid >> 2;      // wave col-tile (64 cols each)
  const int lm   = lane & 15;
  const int hh   = lane >> 4;     // lane half

  const long long blockM = (long long)blockIdx.x * TM;
  const long long blockO = (long long)blockIdx.y * TO;

  // loader mapping: 2 threads per tile row, each owns a 64B half-row
  const int lrow = tid >> 1;            // 0..127
  const int lcol = (tid & 1) << 5;      // 0 or 32 (bf16 units)
  const unsigned gbaseA = (unsigned)(((blockM + lrow) * (long long)N + lcol) * 2);
  const unsigned gbaseB = (unsigned)(((blockO + lrow) * (long long)N + lcol) * 2);
  const unsigned ldsA0 = (unsigned)(unsigned long long)(&sA[0][lrow * LDSS + lcol]);
  const unsigned ldsA1 = (unsigned)(unsigned long long)(&sA[1][lrow * LDSS + lcol]);
  const unsigned ldsB0 = (unsigned)(unsigned long long)(&sB[0][lrow * LDSS + lcol]);
  const unsigned ldsB1 = (unsigned)(unsigned long long)(&sB[1][lrow * LDSS + lcol]);

  v8f acc[2][4];
#pragma unroll
  for (int i = 0; i < 2; ++i)
#pragma unroll
    for (int j = 0; j < 4; ++j) acc[i][j] = (v8f)(0.0f);

  // prologue: stage tile 0 into buffer 0
  async_half_row(ldsA0, gbaseA, xb);
  async_half_row(ldsB0, gbaseB, wb);

  int buf = 0;
  for (int kt = 0; kt < N; kt += TK) {
    wait_async_all();       // my async copies for `buf` have landed in LDS
    __syncthreads();        // everyone's copies have landed

    int ktn = kt + TK;
    if (ktn < N) {          // kick next tile into the other buffer (overlaps wmma)
      unsigned go = (unsigned)(ktn * 2);
      async_half_row(buf ? ldsA0 : ldsA1, gbaseA + go, xb);
      async_half_row(buf ? ldsB0 : ldsB1, gbaseB + go, wb);
    }

    const __bf16* pA = sA[buf];
    const __bf16* pB = sB[buf];
#pragma unroll
    for (int kk = 0; kk < TK; kk += 32) {
      union V16 { v16bf v; v8bf h[2]; };
      V16 a[2], b[4];
      // A frag: lanes 0-15 carry K 0-7/16-23, lanes 16-31 carry K 8-15/24-31
#pragma unroll
      for (int ms = 0; ms < 2; ++ms) {
        int row = wm * 32 + ms * 16 + lm;
        a[ms].h[0] = *(const v8bf*)&pA[row * LDSS + kk + hh * 8];
        a[ms].h[1] = *(const v8bf*)&pA[row * LDSS + kk + 16 + hh * 8];
      }
      // B frag: lane = output column; lanes 0-15 carry K 0-7/16-23, 16-31 K 8-15/24-31
#pragma unroll
      for (int os = 0; os < 4; ++os) {
        int row = wo * 64 + os * 16 + lm;
        b[os].h[0] = *(const v8bf*)&pB[row * LDSS + kk + hh * 8];
        b[os].h[1] = *(const v8bf*)&pB[row * LDSS + kk + 16 + hh * 8];
      }
#pragma unroll
      for (int ms = 0; ms < 2; ++ms)
#pragma unroll
        for (int os = 0; os < 4; ++os)
          acc[ms][os] = __builtin_amdgcn_wmma_f32_16x16x32_bf16(
              false, a[ms].v, false, b[os].v, (short)0, acc[ms][os], false, false);
    }
    buf ^= 1;
    // next iteration's wait+barrier fences both the new async tile and the
    // re-use of this buffer, so no trailing barrier needed here
  }

  // Epilogue: C/D layout -> lane holds column (lm), VGPR i -> row i + 8*hh
#pragma unroll
  for (int os = 0; os < 4; ++os) {
    long long o = blockO + wo * 64 + os * 16 + lm;
    float bv = bias[o];
#pragma unroll
    for (int ms = 0; ms < 2; ++ms) {
      long long mbase = blockM + wm * 32 + ms * 16 + hh * 8;
#pragma unroll
      for (int i = 0; i < 8; ++i)
        out[(mbase + i) * Kout + o] = acc[ms][os][i] + bv;
    }
  }
}

// ---------------- launch ----------------
extern "C" void kernel_launch(void* const* d_in, const int* in_sizes, int n_in,
                              void* d_out, int out_size, void* d_ws, size_t ws_size,
                              hipStream_t stream) {
  const float* x      = (const float*)d_in[0];
  const int*   Wq     = (const int*)d_in[1];
  const float* scales = (const float*)d_in[2];
  const float* zeros  = (const float*)d_in[3];
  const int*   mask   = (const int*)d_in[4];
  const float* scale2 = (const float*)d_in[5];
  const float* bias   = (const float*)d_in[6];
  float* out = (float*)d_out;

  const int Kout = in_sizes[6];                       // 4096
  const long long KN = (long long)in_sizes[1];        // K*N
  const int N = (int)(KN / Kout);                     // 4096
  const long long MN = (long long)in_sizes[0];        // M*N
  const int M = (int)(MN / N);                        // 8192
  const int G = N / GROUPSZ;

  __bf16* wbf = (__bf16*)d_ws;                                              // 32 MiB
  __bf16* xbf = (__bf16*)((char*)d_ws + (size_t)Kout * N * sizeof(__bf16)); // +64 MiB

  {
    long long total8 = KN >> 3;
    int blocks = (int)((total8 + 255) / 256);
    dequant_w_bf16<<<blocks, 256, 0, stream>>>(Wq, scales, zeros, mask, scale2,
                                               wbf, N, G, total8);
  }
  {
    long long total8 = MN >> 3;
    int blocks = (int)((total8 + 255) / 256);
    cvt_x_bf16<<<blocks, 256, 0, stream>>>(x, xbf, total8);
  }
  {
    dim3 grid(M / TM, Kout / TO);   // 64 x 32 = 2048 blocks
    gemm_bf16_wmma<<<grid, 256, 0, stream>>>(xbf, wbf, bias, out, N, Kout);
  }
}
